// QuestionEmbedding_9225589752652
// MI455X (gfx1250) — matrix operand order repaired and checked by reading, not compiled, not run on hardware
//
#include <hip/hip_runtime.h>
#include <hip/hip_bf16.h>
#include <math.h>

typedef __bf16 bf16_t;
typedef __attribute__((ext_vector_type(16))) __bf16 v16bf;
typedef __attribute__((ext_vector_type(8)))  __bf16 v8bf;
typedef __attribute__((ext_vector_type(8)))  float  v8f;

#define B_   2048
#define T_   20
#define IN_  600
#define KXP  640             // x / W_ih K padded to multiple of 64
#define H_   1024
#define G3_  (3 * H_)

#define BK   64              // K-chunk staged per barrier (2 WMMA K-steps)
#define BKP  72              // padded LDS row stride in halves (144B, 16B-aligned)
#define TSZ  (64 * BKP)      // halves per 64-row tile

// ---------------------------------------------------------------------------
// Fragment loaders matching CDNA5 WMMA bf16 lane layouts (wave32).
// A 16x32 (MxK): lane L -> row M = L&15, half = L>>4.
//   element e<8  : K = half*8 + e ; e>=8 : K = 16 + half*8 + (e-8)
// B 32x16 (KxN) staged as Bs[n][k] (weights are row-major [N][K]):
//   lane L -> col N = L&15, half = L>>4; same K order as A
// ---------------------------------------------------------------------------
static __device__ __forceinline__ v16bf load_frag_A(const bf16_t* rowp, int hlf) {
  const v8bf lo = *(const v8bf*)(rowp + hlf * 8);
  const v8bf hi = *(const v8bf*)(rowp + 16 + hlf * 8);
  v16bf a;
#pragma unroll
  for (int i = 0; i < 8; ++i) { a[i] = lo[i]; a[i + 8] = hi[i]; }
  return a;
}

static __device__ __forceinline__ v16bf load_frag_B(const bf16_t* rowp, int hlf) {
  const v8bf lo = *(const v8bf*)(rowp + hlf * 8);
  const v8bf hi = *(const v8bf*)(rowp + 16 + hlf * 8);
  v16bf b;
#pragma unroll
  for (int i = 0; i < 8; ++i) { b[i] = lo[i]; b[i + 8] = hi[i]; }
  return b;
}

// Stage one 64xBK bf16 tile: 16B global loads -> 16B LDS stores.
static __device__ __forceinline__ void stage_tile(bf16_t* __restrict__ dst,
                                                  const bf16_t* __restrict__ src,
                                                  size_t srcStride, int tid) {
#pragma unroll
  for (int i = 0; i < 4; ++i) {              // 64 rows * 64 halves / (128*8)
    int idx = tid + i * 128;
    int r = idx >> 3, c8 = (idx & 7) * 8;
    *(v8bf*)&dst[r * BKP + c8] = *(const v8bf*)&src[(size_t)r * srcStride + c8];
  }
}

// ---------------------------------------------------------------------------
// f32 [rows,kin] -> bf16 [rows,kout] with zero padding. 2-D grid: y = row.
// ---------------------------------------------------------------------------
__global__ void pad_convert_bf16(const float* __restrict__ in,
                                 bf16_t* __restrict__ out,
                                 int kin, int kout) {
  int k = blockIdx.x * blockDim.x + threadIdx.x;
  int r = blockIdx.y;
  if (k >= kout) return;
  out[(size_t)r * kout + k] = (k < kin) ? (bf16_t)in[(size_t)r * kin + k]
                                        : (bf16_t)0.0f;
}

// ---------------------------------------------------------------------------
// Phase 1: gate_x[m,g] = sum_k x[m,k] * W_ih[g,k] + b_ih[g], m = b*T + t.
// Double-buffered LDS, one barrier per K-chunk.
// ---------------------------------------------------------------------------
__global__ __launch_bounds__(128)
void gemm_gate_x(const bf16_t* __restrict__ xb,    // [B*T, KXP]
                 const bf16_t* __restrict__ Wih,   // [G3, KXP]
                 const float* __restrict__ b_ih,   // [G3]
                 float* __restrict__ gx)           // [B*T, G3]
{
  __shared__ __attribute__((aligned(16))) bf16_t As[2][TSZ];
  __shared__ __attribute__((aligned(16))) bf16_t Bs[2][TSZ];

  const int tid  = threadIdx.x;
  const int lane = tid & 31;
  const int warp = tid >> 5;
  const int hlf  = lane >> 4;
  const int l16  = lane & 15;

  const int rowBase = blockIdx.x * 64;   // over M = B*T
  const int colBase = blockIdx.y * 64;   // over G3
  const int mOffW = (warp >> 1) * 32;
  const int nOffW = (warp & 1) * 32;

  const bf16_t* aSrc = xb  + (size_t)rowBase * KXP;
  const bf16_t* bSrc = Wih + (size_t)colBase * KXP;

  v8f acc[2][2] = {};

  const int NK = KXP / BK;               // 10 chunks
  stage_tile(As[0], aSrc, KXP, tid);
  stage_tile(Bs[0], bSrc, KXP, tid);
  __syncthreads();

  for (int kc = 0; kc < NK; ++kc) {
    const int p = kc & 1;
    if (kc + 1 < NK) {                   // prefetch next chunk into other buffer
      stage_tile(As[1 - p], aSrc + (size_t)(kc + 1) * BK, KXP, tid);
      stage_tile(Bs[1 - p], bSrc + (size_t)(kc + 1) * BK, KXP, tid);
    }
#pragma unroll
    for (int kk = 0; kk < 2; ++kk) {     // two K=32 WMMA steps per chunk
      v16bf afr[2];
#pragma unroll
      for (int mi = 0; mi < 2; ++mi)
        afr[mi] = load_frag_A(&As[p][(mOffW + mi * 16 + l16) * BKP + kk * 32], hlf);
#pragma unroll
      for (int ni = 0; ni < 2; ++ni) {
        v16bf bfr = load_frag_B(&Bs[p][(nOffW + ni * 16 + l16) * BKP + kk * 32], hlf);
#pragma unroll
        for (int mi = 0; mi < 2; ++mi)
          acc[mi][ni] = __builtin_amdgcn_wmma_f32_16x16x32_bf16(
              false, afr[mi], false, bfr, (short)0, acc[mi][ni], false, false);
      }
    }
    __syncthreads();
  }

  // Epilogue: C layout -> M = r + 8*half, N = lane&15. gx row index is m.
#pragma unroll
  for (int mi = 0; mi < 2; ++mi) {
#pragma unroll
    for (int ni = 0; ni < 2; ++ni) {
      const int g = colBase + nOffW + ni * 16 + l16;
      const float bias = b_ih[g];
      const int m0 = rowBase + mOffW + mi * 16 + hlf * 8;
#pragma unroll
      for (int r = 0; r < 8; ++r)
        gx[(size_t)(m0 + r) * G3_ + g] = acc[mi][ni][r] + bias;
    }
  }
}

// ---------------------------------------------------------------------------
// Phase 2 (per timestep, fused): gate_h = h_prev * W_hh^T for all 3 gates
// (shared A tile, 3 B tiles), then GRU elementwise update on accumulators.
// Double-buffered LDS, one barrier per K-chunk.
// ---------------------------------------------------------------------------
__global__ __launch_bounds__(128)
void gru_step(const bf16_t* __restrict__ hb_prev,  // [B, H] bf16
              const float*  __restrict__ h_prev,   // [B, H] f32
              const bf16_t* __restrict__ Whh,      // [G3, H] bf16
              const float*  __restrict__ b_hh,     // [G3]
              const float*  __restrict__ gx,       // [B*T, G3]
              float*        __restrict__ h_next,   // [B, H] f32
              bf16_t*       __restrict__ hb_next,  // [B, H] bf16
              int t)
{
  __shared__ __attribute__((aligned(16))) bf16_t As[2][TSZ];
  __shared__ __attribute__((aligned(16))) bf16_t Bs[2][3][TSZ];

  const int tid  = threadIdx.x;
  const int lane = tid & 31;
  const int warp = tid >> 5;
  const int hlf  = lane >> 4;
  const int l16  = lane & 15;

  const int rowBase = blockIdx.x * 64;   // over B
  const int colBase = blockIdx.y * 64;   // over H
  const int mOffW = (warp >> 1) * 32;
  const int nOffW = (warp & 1) * 32;

  const bf16_t* aSrc = hb_prev + (size_t)rowBase * H_;
  const bf16_t* bSrc0 = Whh + (size_t)(0 * H_ + colBase) * H_;
  const bf16_t* bSrc1 = Whh + (size_t)(1 * H_ + colBase) * H_;
  const bf16_t* bSrc2 = Whh + (size_t)(2 * H_ + colBase) * H_;

  v8f acc[3][2][2] = {};

  const int NK = H_ / BK;                // 16 chunks
  stage_tile(As[0], aSrc, H_, tid);
  stage_tile(Bs[0][0], bSrc0, H_, tid);
  stage_tile(Bs[0][1], bSrc1, H_, tid);
  stage_tile(Bs[0][2], bSrc2, H_, tid);
  __syncthreads();

  for (int kc = 0; kc < NK; ++kc) {
    const int p = kc & 1;
    if (kc + 1 < NK) {
      const size_t o = (size_t)(kc + 1) * BK;
      stage_tile(As[1 - p], aSrc + o, H_, tid);
      stage_tile(Bs[1 - p][0], bSrc0 + o, H_, tid);
      stage_tile(Bs[1 - p][1], bSrc1 + o, H_, tid);
      stage_tile(Bs[1 - p][2], bSrc2 + o, H_, tid);
    }
#pragma unroll
    for (int kk = 0; kk < 2; ++kk) {
      v16bf afr[2];
#pragma unroll
      for (int mi = 0; mi < 2; ++mi)
        afr[mi] = load_frag_A(&As[p][(mOffW + mi * 16 + l16) * BKP + kk * 32], hlf);
#pragma unroll
      for (int q = 0; q < 3; ++q) {
#pragma unroll
        for (int ni = 0; ni < 2; ++ni) {
          v16bf bfr =
              load_frag_B(&Bs[p][q][(nOffW + ni * 16 + l16) * BKP + kk * 32], hlf);
#pragma unroll
          for (int mi = 0; mi < 2; ++mi)
            acc[q][mi][ni] = __builtin_amdgcn_wmma_f32_16x16x32_bf16(
                false, afr[mi], false, bfr, (short)0, acc[q][mi][ni], false, false);
        }
      }
    }
    __syncthreads();
  }

  // Fused GRU update on accumulators. gx row for batch b is m = b*T + t.
#pragma unroll
  for (int mi = 0; mi < 2; ++mi) {
#pragma unroll
    for (int ni = 0; ni < 2; ++ni) {
      const int j = colBase + nOffW + ni * 16 + l16;
      const float bhr = b_hh[j];
      const float bhi = b_hh[H_ + j];
      const float bhn = b_hh[2 * H_ + j];
      const int b0 = rowBase + mOffW + mi * 16 + hlf * 8;
#pragma unroll
      for (int r = 0; r < 8; ++r) {
        const int b = b0 + r;
        const float* gb = gx + ((size_t)b * T_ + t) * G3_;
        float hr = acc[0][mi][ni][r] + bhr;
        float hi = acc[1][mi][ni][r] + bhi;
        float hn = acc[2][mi][ni][r] + bhn;
        float ir = gb[j], ii = gb[H_ + j], in_ = gb[2 * H_ + j];
        float resetg = 1.0f / (1.0f + __expf(-(ir + hr)));
        float inputg = 1.0f / (1.0f + __expf(-(ii + hi)));
        float newg   = tanhf(in_ + resetg * hn);
        float hold   = h_prev[(size_t)b * H_ + j];
        float hy     = newg + inputg * (hold - newg);
        h_next[(size_t)b * H_ + j]  = hy;
        hb_next[(size_t)b * H_ + j] = (bf16_t)hy;
      }
    }
  }
}

// ---------------------------------------------------------------------------
extern "C" void kernel_launch(void* const* d_in, const int* in_sizes, int n_in,
                              void* d_out, int out_size, void* d_ws, size_t ws_size,
                              hipStream_t stream) {
  const float* x    = (const float*)d_in[0];
  const float* W_ih = (const float*)d_in[1];
  const float* b_ih = (const float*)d_in[2];
  const float* W_hh = (const float*)d_in[3];
  const float* b_hh = (const float*)d_in[4];
  float* out = (float*)d_out;

  char* ws = (char*)d_ws;
  size_t off = 0;
  auto alloc = [&](size_t bytes) -> void* {
    void* p = ws + off;
    off = (off + bytes + 255) & ~(size_t)255;
    return p;
  };
  const int M1 = B_ * T_;
  float*  gx     = (float*) alloc((size_t)M1 * G3_ * sizeof(float));
  bf16_t* xb     = (bf16_t*)alloc((size_t)M1 * KXP * sizeof(bf16_t));
  bf16_t* Wih_b  = (bf16_t*)alloc((size_t)G3_ * KXP * sizeof(bf16_t));
  bf16_t* Whh_b  = (bf16_t*)alloc((size_t)G3_ * H_ * sizeof(bf16_t));
  float*  h0f    = (float*) alloc((size_t)B_ * H_ * sizeof(float));
  float*  h1f    = (float*) alloc((size_t)B_ * H_ * sizeof(float));
  bf16_t* h0b    = (bf16_t*)alloc((size_t)B_ * H_ * sizeof(bf16_t));
  bf16_t* h1b    = (bf16_t*)alloc((size_t)B_ * H_ * sizeof(bf16_t));

  // Prep: pad/convert operands to bf16 (deterministic, re-done each call).
  {
    dim3 g((KXP + 255) / 256, M1);
    pad_convert_bf16<<<g, 256, 0, stream>>>(x, xb, IN_, KXP);
  }
  {
    dim3 g((KXP + 255) / 256, G3_);
    pad_convert_bf16<<<g, 256, 0, stream>>>(W_ih, Wih_b, IN_, KXP);
  }
  {
    dim3 g((H_ + 255) / 256, G3_);
    pad_convert_bf16<<<g, 256, 0, stream>>>(W_hh, Whh_b, H_, H_);
  }

  // h0 = zeros (both precisions).
  hipMemsetAsync(h0f, 0, (size_t)B_ * H_ * sizeof(float), stream);
  hipMemsetAsync(h0b, 0, (size_t)B_ * H_ * sizeof(bf16_t), stream);

  // Phase 1: gate_x for all timesteps.
  dim3 g1(M1 / 64, G3_ / 64);
  gemm_gate_x<<<g1, 128, 0, stream>>>(xb, Wih_b, b_ih, gx);

  // Phase 2: 20 sequential fused GEMM + GRU-update steps, ping-pong h.
  dim3 g2(B_ / 64, H_ / 64);
  float*  curf = h0f; bf16_t* curb = h0b;
  float*  nxtf = h1f; bf16_t* nxtb = h1b;
  for (int t = 0; t < T_; ++t) {
    float* of = (t == T_ - 1) ? out : nxtf;
    gru_step<<<g2, 128, 0, stream>>>(curb, curf, Whh_b, b_hh, gx, of, nxtb, t);
    float*  tf = curf; curf = of;   nxtf = tf;
    bf16_t* tb = curb; curb = nxtb; nxtb = tb;
  }
}
